// MultiHeadAttention_29652454211677
// MI455X (gfx1250) — matrix operand order repaired and checked by reading, not compiled, not run on hardware
//
#include <hip/hip_runtime.h>
#include <hip/hip_bf16.h>

typedef __attribute__((ext_vector_type(16))) __bf16 v16bf;
typedef __attribute__((ext_vector_type(8)))  __bf16 v8bf;
typedef __attribute__((ext_vector_type(4)))  __bf16 v4bf;
typedef __attribute__((ext_vector_type(8)))  float  v8f;

#define HID_ 1024
#define T_   2048
#define N_   2
#define H_   16
#define D_   64
#define M_   (N_ * T_)   // 4096 rows for the projection GEMMs

// ---- gfx1250 async global->LDS path (ASYNCcnt) ----------------------------
// Probe-confirmed present; signature wants int4 pointers: (global, lds, imm, imm)
#if __has_builtin(__builtin_amdgcn_global_load_async_to_lds_b128)
#define HAVE_ASYNC_LDS 1
typedef __attribute__((ext_vector_type(4))) int v4i;
typedef __attribute__((address_space(1))) v4i as1_v4i;
typedef __attribute__((address_space(3))) v4i as3_v4i;
#endif

__device__ __forceinline__ void wait_asynccnt0()
{
#if defined(HAVE_ASYNC_LDS)
#if __has_builtin(__builtin_amdgcn_s_wait_asynccnt)
    __builtin_amdgcn_s_wait_asynccnt(0);
#else
    asm volatile("s_wait_asynccnt 0" ::: "memory");
#endif
#endif
}

// ---------------------------------------------------------------------------
// f32 -> bf16 pre-convert (vectorized), so GEMM A-operands are raw bf16 and
// the A tile staging is a pure byte copy (async-to-LDS eligible).
// ---------------------------------------------------------------------------
__global__ __launch_bounds__(256)
void cvt_f32_bf16_kernel(const float* __restrict__ src, __bf16* __restrict__ dst)
{
    int i = blockIdx.x * 256 + threadIdx.x;     // one float4 per thread
    float4 f = ((const float4*)src)[i];
    v4bf o;
    o[0] = (__bf16)f.x; o[1] = (__bf16)f.y; o[2] = (__bf16)f.z; o[3] = (__bf16)f.w;
    ((v4bf*)dst)[i] = o;
}

// ---------------------------------------------------------------------------
// GEMM:  Y[M,Nn] = A_bf16[M,K] @ W_f32[K,Nn] + bias   (Y bf16 or f32)
// 128x128 tile per workgroup, 8 waves (4x2 grid of 32x64 wave tiles),
// BK = 32 == WMMA K.  A tile staged via async-to-LDS when available.
// ---------------------------------------------------------------------------
template <bool OUT_F32>
__global__ __launch_bounds__(256)
void gemm_bias_kernel(const __bf16* __restrict__ Ab,
                      const float* __restrict__ W,
                      const float* __restrict__ bias,
                      __bf16* __restrict__ Ybf,
                      float*  __restrict__ Yf,
                      int M, int K, int Nn)
{
    __shared__ __align__(16) __bf16 As[128 * 32];   // [m][k]
    __shared__ __align__(16) __bf16 Bs[128 * 32];   // [n][k]  (transposed)

    const int tid   = threadIdx.x;
    const int lane  = tid & 31;
    const int wid   = tid >> 5;
    const int waveM = wid >> 1;       // 0..3  -> 32-row slab
    const int waveN = wid & 1;        // 0..1  -> 64-col slab
    const int m0    = blockIdx.y * 128;
    const int n0    = blockIdx.x * 128;

    v8f acc[2][4] = {};               // 2 m-tiles x 4 n-tiles of 16x16

    const int mrow  = lane & 15;
    const int kbA   = (lane >> 4) << 3;   // A frag: upper half-wave K base +8
    const int kbB   = (lane >> 4) << 3;   // B frag: upper half-wave K base +8
    const int rowHi = (lane >> 4) << 3;   // C/D: upper half-wave rows +8

    for (int k0 = 0; k0 < K; k0 += 32) {
        // ---- stage A tile (128x32 bf16 = 8KB): raw copy, 16B per thread ----
#if defined(HAVE_ASYNC_LDS)
        #pragma unroll
        for (int r = 0; r < 2; ++r) {
            int idx = tid + r * 256;              // 0..511
            int mm = idx >> 2, c = idx & 3;       // row, 16B chunk
            const __bf16* gp = &Ab[(size_t)(m0 + mm) * K + k0 + c * 8];
            __bf16* lp = &As[mm * 32 + c * 8];
            __builtin_amdgcn_global_load_async_to_lds_b128(
                (as1_v4i*)gp, (as3_v4i*)lp, 0, 0);
        }
#else
        #pragma unroll
        for (int r = 0; r < 2; ++r) {
            int idx = tid + r * 256;
            int mm = idx >> 2, c = idx & 3;
            *(v8bf*)&As[mm * 32 + c * 8] =
                *(const v8bf*)&Ab[(size_t)(m0 + mm) * K + k0 + c * 8];
        }
#endif
        // ---- stage B tile (32x128) transposed+converted -> Bs[n][k] ----
        #pragma unroll
        for (int r = 0; r < 16; ++r) {
            int idx = tid + r * 256;
            int kk = idx >> 7, nn = idx & 127;
            Bs[nn * 32 + kk] = (__bf16)W[(size_t)(k0 + kk) * Nn + n0 + nn];
        }
        // prefetch next K-tile (gfx1250 global_prefetch_b8 path)
        if (k0 + 32 < K) {
            __builtin_prefetch(&Ab[(size_t)(m0 + (tid >> 1)) * K + k0 + 32], 0, 1);
            __builtin_prefetch(&W[(size_t)(k0 + 32 + (tid >> 3)) * Nn + n0 + (tid & 7) * 16], 0, 1);
        }
        wait_asynccnt0();
        __syncthreads();

        // ---- gather WMMA fragments from LDS ----
        v16bf afrag[2], bfrag[4];
        #pragma unroll
        for (int mt = 0; mt < 2; ++mt) {
            const __bf16* ap = &As[(waveM * 32 + mt * 16 + mrow) * 32];
            #pragma unroll
            for (int v = 0; v < 8; ++v) {
                int kk = ((v & 3) << 1) + ((v >> 2) << 4) + kbA;
                afrag[mt][2 * v]     = ap[kk];
                afrag[mt][2 * v + 1] = ap[kk + 1];
            }
        }
        #pragma unroll
        for (int nt = 0; nt < 4; ++nt) {
            const __bf16* bp = &Bs[(waveN * 64 + nt * 16 + mrow) * 32];
            #pragma unroll
            for (int v = 0; v < 8; ++v) {
                int kk = ((v & 3) << 1) + ((v >> 2) << 4) + kbB;
                bfrag[nt][2 * v]     = bp[kk];
                bfrag[nt][2 * v + 1] = bp[kk + 1];
            }
        }
        // ---- 8 WMMAs per wave per K-step ----
        #pragma unroll
        for (int mt = 0; mt < 2; ++mt)
            #pragma unroll
            for (int nt = 0; nt < 4; ++nt)
                acc[mt][nt] = __builtin_amdgcn_wmma_f32_16x16x32_bf16(
                    false, afrag[mt], false, bfrag[nt],
                    (short)0, acc[mt][nt], false, false);
        __syncthreads();
    }

    // ---- epilogue: bias + store ----
    #pragma unroll
    for (int mt = 0; mt < 2; ++mt) {
        #pragma unroll
        for (int nt = 0; nt < 4; ++nt) {
            int nc = n0 + waveN * 64 + nt * 16 + mrow;
            float bv = bias[nc];
            #pragma unroll
            for (int v = 0; v < 8; ++v) {
                int mr = m0 + waveM * 32 + mt * 16 + rowHi + v;
                float val = acc[mt][nt][v] + bv;
                if (OUT_F32) Yf[(size_t)mr * Nn + nc] = val;
                else         Ybf[(size_t)mr * Nn + nc] = (__bf16)val;
            }
        }
    }
}

// ---------------------------------------------------------------------------
// Attention with softmax over the HEAD axis.
// Block = 512 threads = 16 waves, one wave per head; one 16-row t-tile/block.
// Per 32-wide s-tile: V tile staged transposed into LDS (coalesced v8bf reads,
// contiguous-pair ds reads for B fragments), S_h = Q_h @ K_h^T (4 WMMA),
// cross-head exp/sum in LDS, then O_h += A_h @ V_h (4 WMMA).
// ---------------------------------------------------------------------------
__global__ __launch_bounds__(512)
void attention_kernel(const __bf16* __restrict__ Qp,
                      const __bf16* __restrict__ Kp,
                      const __bf16* __restrict__ Vp,
                      __bf16* __restrict__ Ctx)
{
    __shared__ float  Ssc[H_ * 16 * 32];          // [h][t][s]   32 KB
    __shared__ float  rden[16 * 32];              // [t][s]       2 KB
    __shared__ __align__(16) __bf16 Vt[H_ * D_ * 32]; // [h][d][s] 64 KB

    const int tid  = threadIdx.x;
    const int lane = tid & 31;
    const int h    = tid >> 5;            // wave id == head
    const int n    = blockIdx.y;
    const int t0   = blockIdx.x * 16;

    const size_t headBase = ((size_t)(n * H_ + h)) * T_ * D_;   // (N,H,T,D) flat

    const int mrow  = lane & 15;
    const int kbA   = (lane >> 4) << 3;
    const int kbB   = (lane >> 4) << 3;
    const int rowHi = (lane >> 4) << 3;
    const float scale = 0.03125f;         // 1/sqrt(1024)

    // ---- Q fragments for this wave's head (16 x 64 -> two 16x32 frags) ----
    v16bf qa[2];
    {
        const __bf16* qrow = Qp + headBase + (size_t)(t0 + mrow) * D_;
        #pragma unroll
        for (int c = 0; c < 2; ++c)
            #pragma unroll
            for (int v = 0; v < 8; ++v) {
                int d = c * 32 + ((v & 3) << 1) + ((v >> 2) << 4) + kbA;
                qa[c][2 * v]     = qrow[d];
                qa[c][2 * v + 1] = qrow[d + 1];
            }
    }

    v8f oacc[4] = {};   // O tile 16x64 f32

    for (int s0 = 0; s0 < T_; s0 += 32) {
        // ---- stage this head's V tile (32 s x 64 d) transposed -> Vt[h][d][s]
        {
            const __bf16* vbase = Vp + headBase + (size_t)s0 * D_;
            #pragma unroll
            for (int r = 0; r < 8; ++r) {
                int idx = lane + r * 32;     // 0..255 (x8 bf16 = 2048 elems)
                int s  = idx >> 3;           // 0..31
                int d8 = (idx & 7) * 8;      // 0,8,...,56
                v8bf vv = *(const v8bf*)(vbase + (size_t)s * D_ + d8);
                #pragma unroll
                for (int e = 0; e < 8; ++e)
                    Vt[h * 2048 + (d8 + e) * 32 + s] = vv[e];
            }
        }
        // ---- scores: two 16x16 tiles, K = 64 over two WMMAs each ----
        v8f sc[2] = {};
        #pragma unroll
        for (int st = 0; st < 2; ++st) {
            const __bf16* krow = Kp + headBase + (size_t)(s0 + st * 16 + mrow) * D_;
            #pragma unroll
            for (int c = 0; c < 2; ++c) {
                v16bf kf;   // B = K^T : N=s (lane), K=d
                #pragma unroll
                for (int v = 0; v < 8; ++v) {
                    int d = c * 32 + ((v & 3) << 1) + ((v >> 2) << 4) + kbB;
                    kf[2 * v]     = krow[d];
                    kf[2 * v + 1] = krow[d + 1];
                }
                sc[st] = __builtin_amdgcn_wmma_f32_16x16x32_bf16(
                    false, qa[c], false, kf, (short)0, sc[st], false, false);
            }
        }
        // ---- scaled scores -> LDS [h][t][s] ----
        #pragma unroll
        for (int st = 0; st < 2; ++st)
            #pragma unroll
            for (int v = 0; v < 8; ++v)
                Ssc[h * 512 + (rowHi + v) * 32 + st * 16 + mrow] = sc[st][v] * scale;
        __syncthreads();

        // ---- softmax across heads: each of 512 threads owns one (t,s) ----
        {
            int t = tid >> 5;
            int s = tid & 31;
            float sum = 0.f;
            #pragma unroll
            for (int hh = 0; hh < H_; ++hh) {
                float e = __expf(Ssc[hh * 512 + t * 32 + s]);
                Ssc[hh * 512 + t * 32 + s] = e;
                sum += e;
            }
            rden[t * 32 + s] = 1.0f / sum;
        }
        __syncthreads();

        // ---- A fragment: normalized attn (16 t x 32 s) as bf16 ----
        v16bf af;
        #pragma unroll
        for (int v = 0; v < 8; ++v) {
            int s = ((v & 3) << 1) + ((v >> 2) << 4) + kbA;
            float r0 = rden[mrow * 32 + s];
            float r1 = rden[mrow * 32 + s + 1];
            af[2 * v]     = (__bf16)(Ssc[h * 512 + mrow * 32 + s]     * r0);
            af[2 * v + 1] = (__bf16)(Ssc[h * 512 + mrow * 32 + s + 1] * r1);
        }

        // ---- O(16x64) += A(16x32) @ V(32x64): 4 WMMAs over d-subtiles ----
        #pragma unroll
        for (int dt = 0; dt < 4; ++dt) {
            v16bf vf;   // B = V : N=d (lane), K=s ; contiguous pairs in Vt
            const __bf16* vp2 = &Vt[h * 2048 + (dt * 16 + mrow) * 32 + kbB];
            #pragma unroll
            for (int v = 0; v < 8; ++v) {
                vf[2 * v]     = vp2[((v & 3) << 1) + ((v >> 2) << 4)];
                vf[2 * v + 1] = vp2[((v & 3) << 1) + ((v >> 2) << 4) + 1];
            }
            oacc[dt] = __builtin_amdgcn_wmma_f32_16x16x32_bf16(
                false, af, false, vf, (short)0, oacc[dt], false, false);
        }
        __syncthreads();   // LDS reused next s-iteration
    }

    // ---- store context tile: (N,H,T,D) flat == (N,T,HID) reshape ----
    #pragma unroll
    for (int dt = 0; dt < 4; ++dt)
        #pragma unroll
        for (int v = 0; v < 8; ++v) {
            int t = t0 + rowHi + v;
            int d = dt * 16 + mrow;
            Ctx[headBase + (size_t)t * D_ + d] = (__bf16)oacc[dt][v];
        }
}

// ---------------------------------------------------------------------------
extern "C" void kernel_launch(void* const* d_in, const int* in_sizes, int n_in,
                              void* d_out, int out_size, void* d_ws, size_t ws_size,
                              hipStream_t stream)
{
    const float* q  = (const float*)d_in[0];
    const float* k  = (const float*)d_in[1];
    const float* v  = (const float*)d_in[2];
    const float* Wq = (const float*)d_in[3];
    const float* bq = (const float*)d_in[4];
    const float* Wk = (const float*)d_in[5];
    const float* bk = (const float*)d_in[6];
    const float* Wv = (const float*)d_in[7];
    const float* bv = (const float*)d_in[8];
    const float* Wo = (const float*)d_in[9];
    const float* bo = (const float*)d_in[10];
    float* out = (float*)d_out;

    const size_t EL = (size_t)N_ * T_ * HID_;   // 4,194,304 elements
    __bf16* Qp = (__bf16*)d_ws;
    __bf16* Kp = Qp + EL;
    __bf16* Vp = Kp + EL;
    __bf16* X  = Vp + EL;    // scratch slab: cvt buffer, then attention context

    dim3 cblk(256), cgrid(EL / 1024);           // 4 elems/thread
    dim3 gblk(256), ggrid(HID_ / 128, M_ / 128);

    // Q/K/V projections: convert activations to bf16, then WMMA GEMM
    cvt_f32_bf16_kernel<<<cgrid, cblk, 0, stream>>>(q, X);
    gemm_bias_kernel<false><<<ggrid, gblk, 0, stream>>>(X, Wq, bq, Qp, nullptr, M_, HID_, HID_);
    cvt_f32_bf16_kernel<<<cgrid, cblk, 0, stream>>>(k, X);
    gemm_bias_kernel<false><<<ggrid, gblk, 0, stream>>>(X, Wk, bk, Kp, nullptr, M_, HID_, HID_);
    cvt_f32_bf16_kernel<<<cgrid, cblk, 0, stream>>>(v, X);
    gemm_bias_kernel<false><<<ggrid, gblk, 0, stream>>>(X, Wv, bv, Vp, nullptr, M_, HID_, HID_);

    // attention with head-axis softmax; context written into X
    attention_kernel<<<dim3(T_ / 16, N_), 512, 0, stream>>>(Qp, Kp, Vp, X);

    // output projection (bf16 context, f32 out)
    gemm_bias_kernel<true><<<ggrid, gblk, 0, stream>>>(X, Wo, bo, nullptr, out, M_, HID_, HID_);
}
